// LSTMLayer_2972117369275
// MI455X (gfx1250) — matrix operand order repaired and checked by reading, not compiled, not run on hardware
//
#include <hip/hip_runtime.h>
#include <hip/hip_bf16.h>

// ---------------------------------------------------------------------------
// LSTM layer for MI455X (gfx1250, wave32, WMMA).
//   T=1024, B=64, I=H=512.  Phase 1: gx = x @ w_ih^T + b  (big bf16 WMMA GEMM)
//   Phase 2: persistent-kernel recurrence, w_hh resident in LDS, h ping-pong
//            in global bf16 buffers, grid barrier via atomics per step.
// ---------------------------------------------------------------------------

#define TT 1024
#define BB 64
#define II 512
#define HH 512
#define GH 2048              // 4*H
#define NWG_REC 64           // persistent workgroups for recurrence
#define HCOLS 8              // hidden columns owned per recurrent WG

typedef __bf16 bf16;
typedef __attribute__((ext_vector_type(8)))  __bf16 v8bf;
typedef __attribute__((ext_vector_type(16))) __bf16 v16bf;
typedef __attribute__((ext_vector_type(8)))  float  v8f;

__device__ inline v16bf cat16(v8bf lo, v8bf hi) {
    return __builtin_shufflevector(lo, hi, 0,1,2,3,4,5,6,7,8,9,10,11,12,13,14,15);
}
__device__ inline float sigf(float x) { return 1.0f / (1.0f + __expf(-x)); }

// ---------------- workspace layout (bytes) ----------------
#define OFF_XB   ((size_t)0)                       // T*B*I bf16   = 64 MB
#define OFF_WIH  ((size_t)67108864)                // GH*I bf16    = 2 MB
#define OFF_GX   ((size_t)69206016)                // T*B*GH f32   = 512 MB
#define OFF_H0   ((size_t)606076928)               // B*H bf16     = 64 KB
#define OFF_H1   ((size_t)606142464)               // B*H bf16     = 64 KB
#define OFF_BAR  ((size_t)606208000)               // int

// ---------------------------------------------------------------------------
// Kernel 0: fp32 -> bf16 conversions + h0*mask + barrier reset (every launch,
// so graph replay stays deterministic).
// ---------------------------------------------------------------------------
__global__ __launch_bounds__(256) void lstm_init_kernel(
    const float* __restrict__ x, const float* __restrict__ w_ih,
    const float* __restrict__ h0, const float* __restrict__ mask,
    bf16* __restrict__ xb, bf16* __restrict__ wihb,
    bf16* __restrict__ hbuf0, int* __restrict__ bar)
{
    size_t gid    = (size_t)blockIdx.x * blockDim.x + threadIdx.x;
    size_t stride = (size_t)gridDim.x * blockDim.x;
    for (size_t i = gid; i < (size_t)TT * BB * II; i += stride)
        xb[i] = (bf16)x[i];
    for (size_t i = gid; i < (size_t)GH * II; i += stride)
        wihb[i] = (bf16)w_ih[i];
    for (size_t i = gid; i < (size_t)BB * HH; i += stride)
        hbuf0[i] = (bf16)(h0[i] * mask[i]);
    if (gid == 0) *bar = 0;
}

// ---------------------------------------------------------------------------
// Kernel 1: gx[t*B+b, g] = sum_k x[t,b,k] * w_ih[g,k] + b_ih[g] + b_hh[g]
// M = T*B = 65536, N = 2048, K = 512. bf16 WMMA, f32 accumulate.
// Block = 256 threads (8 waves as 2x4), WG tile 128M x 128N, wave tile 64x32.
// ---------------------------------------------------------------------------
__global__ __launch_bounds__(256) void gx_gemm_kernel(
    const bf16* __restrict__ xb, const bf16* __restrict__ wihb,
    const float* __restrict__ b_ih, const float* __restrict__ b_hh,
    float* __restrict__ gx)
{
    const int wave = threadIdx.x >> 5;
    const int lane = threadIdx.x & 31;
    const int lr   = lane & 15;
    const int half = lane >> 4;
    const int wm   = wave >> 2;          // 0..1
    const int wn   = wave & 3;           // 0..3
    const int bm   = blockIdx.y * 128 + wm * 64;
    const int bn   = blockIdx.x * 128 + wn * 32;

    v8f acc[4][2] = {};

    for (int kk = 0; kk < 16; ++kk) {
        const int k0 = kk * 32;
        v16bf a[4], b[2];
#pragma unroll
        for (int mi = 0; mi < 4; ++mi) {
            // A (16-bit 16x32): lane halves hold K chunks [k0+8h .. +7] and [k0+16+8h .. +7]
            const bf16* ap = xb + (size_t)(bm + mi * 16 + lr) * II + k0 + half * 8;
            a[mi] = cat16(*(const v8bf*)ap, *(const v8bf*)(ap + 16));
        }
#pragma unroll
        for (int ni = 0; ni < 2; ++ni) {
            // B (32x16): lane = column, halves hold K chunks [k0+8h .. +7] and [k0+16+8h .. +7]
            const bf16* bp = wihb + (size_t)(bn + ni * 16 + lr) * II + k0 + half * 8;
            b[ni] = cat16(*(const v8bf*)bp, *(const v8bf*)(bp + 16));
        }
#pragma unroll
        for (int mi = 0; mi < 4; ++mi)
#pragma unroll
            for (int ni = 0; ni < 2; ++ni)
                acc[mi][ni] = __builtin_amdgcn_wmma_f32_16x16x32_bf16(
                    false, a[mi], false, b[ni], (short)0, acc[mi][ni], false, false);
    }

#pragma unroll
    for (int mi = 0; mi < 4; ++mi)
#pragma unroll
        for (int ni = 0; ni < 2; ++ni) {
            const int col  = bn + ni * 16 + lr;
            const float bi = b_ih[col] + b_hh[col];
#pragma unroll
            for (int r = 0; r < 8; ++r) {
                const int row = bm + mi * 16 + half * 8 + r;  // C layout: VGPR r -> M = r + 8*half
                gx[(size_t)row * GH + col] = acc[mi][ni][r] + bi;
            }
        }
}

// ---------------------------------------------------------------------------
// Kernel 2: persistent recurrence. 64 WGs x 256 threads. WG k owns hidden
// columns [k*8, k*8+8) for all 4 gates -> 32 gate columns, K = 512.
// w_hh slice lives in LDS (bf16, 32KB). One 16x16 WMMA C-tile per wave.
// ---------------------------------------------------------------------------
__device__ inline void grid_sync(int* bar, int target) {
    __syncthreads();
    if (threadIdx.x == 0) {
        __threadfence();
        atomicAdd(bar, 1);
        while (__hip_atomic_load(bar, __ATOMIC_ACQUIRE, __HIP_MEMORY_SCOPE_AGENT) < target) {
            __builtin_amdgcn_s_sleep(1);
        }
    }
    __syncthreads();
}

__global__ __launch_bounds__(256) void lstm_recurrent_kernel(
    const float* __restrict__ gx, const float* __restrict__ w_hh,
    const float* __restrict__ c0, const float* __restrict__ mask,
    bf16* __restrict__ hbuf0, bf16* __restrict__ hbuf1,
    int* __restrict__ bar, float* __restrict__ out)
{
    __shared__ bf16  lds_w[32][II];      // w_hh slice: row = q*8+c, K contiguous (32 KB)
    __shared__ float lds_gates[BB][32];  // gate pre-activations for this WG (8 KB)
    __shared__ float lds_c[BB][HCOLS];   // cell state (2 KB)
    __shared__ float lds_m[BB][HCOLS];   // dropout mask (2 KB)

    const int wg   = blockIdx.x;         // 0..63
    const int tid  = threadIdx.x;
    const int wave = tid >> 5;
    const int lane = tid & 31;
    const int lr   = lane & 15;
    const int half = lane >> 4;
    const int m    = wave >> 1;          // 0..3 : 16-row block of batch
    const int n    = wave & 1;           // 0..1 : 16-col block of the 32 gate cols
    const int sc   = n * 16 + lr;        // slice column 0..31
    const int q    = sc >> 3;            // gate index 0..3
    const int gcol = q * HH + wg * HCOLS + (sc & 7);   // global gate column

    // ---- load w_hh slice into LDS as bf16 ----
    for (int idx = tid; idx < 32 * II; idx += 256) {
        const int r = idx >> 9;          // 0..31
        const int k = idx & 511;
        const int grow = (r >> 3) * HH + wg * HCOLS + (r & 7);
        lds_w[r][k] = (bf16)w_hh[(size_t)grow * HH + k];
    }
    // ---- c0 and mask for owned columns ----
    for (int idx = tid; idx < BB * HCOLS; idx += 256) {
        const int b_ = idx >> 3, c_ = idx & 7;
        const int col = wg * HCOLS + c_;
        lds_c[b_][c_] = c0[b_ * HH + col];
        lds_m[b_][c_] = mask[b_ * HH + col];
    }
    __syncthreads();

    bf16* hbufs[2] = { hbuf0, hbuf1 };

    for (int t = 0; t < TT; ++t) {
        const bf16* hc = hbufs[t & 1];
        bf16*       hn = hbufs[(t + 1) & 1];

        // ---- accumulator initialized from precomputed gx (C layout) ----
        v8f acc;
#pragma unroll
        for (int r = 0; r < 8; ++r) {
            const int row = m * 16 + half * 8 + r;
            acc[r] = gx[((size_t)t * BB + row) * GH + gcol];
        }

        // ---- (h * mask) @ w_hh^T : K = 512 in 16 WMMA steps ----
#pragma unroll 4
        for (int kk = 0; kk < 16; ++kk) {
            const int k0 = kk * 32;
            const bf16* ap = hc + (m * 16 + lr) * HH + k0 + half * 8;
            v16bf a = cat16(*(const v8bf*)ap, *(const v8bf*)(ap + 16));
            const bf16* bp = &lds_w[sc][k0 + half * 8];
            v16bf b = cat16(*(const v8bf*)bp, *(const v8bf*)(bp + 16));
            acc = __builtin_amdgcn_wmma_f32_16x16x32_bf16(
                false, a, false, b, (short)0, acc, false, false);
        }

        // ---- scatter gates to LDS so all 4 gates of a column are visible ----
#pragma unroll
        for (int r = 0; r < 8; ++r)
            lds_gates[m * 16 + half * 8 + r][sc] = acc[r];
        __syncthreads();

        // ---- elementwise LSTM cell update for owned columns ----
        for (int e = tid; e < BB * HCOLS; e += 256) {
            const int b_ = e >> 3, c_ = e & 7;
            const float iv = sigf(lds_gates[b_][c_]);
            const float fv = sigf(lds_gates[b_][8 + c_]);
            const float gv = tanhf(lds_gates[b_][16 + c_]);
            const float ov = sigf(lds_gates[b_][24 + c_]);
            const float cn = fv * lds_c[b_][c_] + iv * gv;
            lds_c[b_][c_] = cn;
            const float hv = ov * tanhf(cn);
            const int col = wg * HCOLS + c_;
            out[(size_t)t * BB * HH + b_ * HH + col] = hv;              // ys
            hn[b_ * HH + col] = (bf16)(hv * lds_m[b_][c_]);             // masked h for next step
            if (t == TT - 1) {
                out[(size_t)TT * BB * HH + b_ * HH + col] = hv;         // hT
                out[(size_t)TT * BB * HH + BB * HH + b_ * HH + col] = cn; // cT
            }
        }

        // ---- device-wide barrier between timesteps ----
        grid_sync(bar, NWG_REC * (t + 1));
    }
}

// ---------------------------------------------------------------------------
extern "C" void kernel_launch(void* const* d_in, const int* in_sizes, int n_in,
                              void* d_out, int out_size, void* d_ws, size_t ws_size,
                              hipStream_t stream)
{
    const float* x     = (const float*)d_in[0];
    const float* h0    = (const float*)d_in[1];
    const float* c0    = (const float*)d_in[2];
    const float* w_ih  = (const float*)d_in[3];
    const float* w_hh  = (const float*)d_in[4];
    const float* b_ih  = (const float*)d_in[5];
    const float* b_hh  = (const float*)d_in[6];
    const float* mask  = (const float*)d_in[7];
    float* out = (float*)d_out;
    char*  ws  = (char*)d_ws;

    bf16* xb     = (bf16*)(ws + OFF_XB);
    bf16* wihb   = (bf16*)(ws + OFF_WIH);
    float* gx    = (float*)(ws + OFF_GX);
    bf16* hbuf0  = (bf16*)(ws + OFF_H0);
    bf16* hbuf1  = (bf16*)(ws + OFF_H1);
    int*  bar    = (int*)(ws + OFF_BAR);

    lstm_init_kernel<<<4096, 256, 0, stream>>>(x, w_ih, h0, mask, xb, wihb, hbuf0, bar);

    gx_gemm_kernel<<<dim3(GH / 128, (TT * BB) / 128), 256, 0, stream>>>(
        xb, wihb, b_ih, b_hh, gx);

    lstm_recurrent_kernel<<<NWG_REC, 256, 0, stream>>>(
        gx, w_hh, c0, mask, hbuf0, hbuf1, bar, out);
}
